// CopyStack_53506702573701
// MI455X (gfx1250) — matrix operand order repaired and checked by reading, not compiled, not run on hardware
//
#include <hip/hip_runtime.h>
#include <hip/hip_bf16.h>
#include <math.h>

typedef __attribute__((ext_vector_type(16))) __bf16 v16bf;
typedef __attribute__((ext_vector_type(8)))  __bf16 v8bf;
typedef __attribute__((ext_vector_type(8)))  float  v8f;

#define B_  4
#define S_  512
#define T_  256
#define H_  1024
#define V_  32000

// GEMM tiling: block tile 128(M) x 64(N), K-step 32, 8 waves as 4(M) x 2(N),
// each wave owns a 32x32 subtile = 2x2 WMMA 16x16 accumulators.
#define BM 128
#define BN 64
#define BK 32
#define LDA 40   // padded LDS row stride (bf16 elems): 80B -> conflict-free b128 rows
#define LDB 40

union Frag { v16bf v; v8bf h[2]; };

__device__ __forceinline__ v8f wmma_bf16(const Frag& a, const Frag& b, v8f c) {
  return __builtin_amdgcn_wmma_f32_16x16x32_bf16(
      /*neg_a=*/false, a.v, /*neg_b=*/false, b.v,
      /*c_mod=*/(short)0, c, /*reuse_a=*/false, /*reuse_b=*/false);
}

// ---------------------------------------------------------------------------
// Kernel 1: enc_bf16[m,n] = bf16(tanh(sum_k encoder[m,k]*W[k,n] + bias[n]))
//           M = B*S = 2048, N = K = H = 1024
// ---------------------------------------------------------------------------
__global__ __launch_bounds__(256) void k_gemm_tanh(
    const float* __restrict__ A,     // [M, H] encoder (B,S flattened)
    const float* __restrict__ W,     // [H, H]
    const float* __restrict__ bias,  // [H]
    __bf16* __restrict__ out)        // [M, H] bf16
{
  __shared__ __bf16 As[BM * LDA];
  __shared__ __bf16 Bt[BN * LDB];   // W tile transposed: Bt[n][k]
  const int tid  = threadIdx.x;
  const int m0   = blockIdx.x * BM;
  const int n0   = blockIdx.y * BN;
  const int wave = tid >> 5, lane = tid & 31;
  const int half = lane >> 4, lm = lane & 15;
  const int wm = (wave & 3) * 32, wn = (wave >> 2) * 32;

  v8f acc[2][2] = {};
  for (int k0 = 0; k0 < H_; k0 += BK) {
    if (k0 + BK < H_)  // prefetch next A tile slab (global_prefetch_b8)
      __builtin_prefetch(A + (size_t)(m0 + (tid >> 1)) * H_ + k0 + BK, 0, 1);

    // --- load phase: batch all global_load_b128 so they overlap ---
    float4 fa[4], fb[2];
#pragma unroll
    for (int j = 0; j < 4; ++j) {      // A tile: 128x32 f32 = 1024 float4 chunks
      int i = tid + j * 256;
      int row = i >> 3, c4 = (i & 7) * 4;
      fa[j] = *(const float4*)(A + (size_t)(m0 + row) * H_ + k0 + c4);
    }
#pragma unroll
    for (int j = 0; j < 2; ++j) {      // W tile: 32x64 f32 = 512 float4 chunks
      int i = tid + j * 256;
      int kk = i >> 4, n4 = (i & 15) * 4;
      fb[j] = *(const float4*)(W + (size_t)(k0 + kk) * H_ + n0 + n4);
    }
    // --- convert + LDS store phase ---
#pragma unroll
    for (int j = 0; j < 4; ++j) {
      int i = tid + j * 256;
      int row = i >> 3, c4 = (i & 7) * 4;
      __bf16* d = &As[row * LDA + c4];
      d[0] = (__bf16)fa[j].x; d[1] = (__bf16)fa[j].y;
      d[2] = (__bf16)fa[j].z; d[3] = (__bf16)fa[j].w;
    }
#pragma unroll
    for (int j = 0; j < 2; ++j) {      // transpose into Bt[n][k]
      int i = tid + j * 256;
      int kk = i >> 4, n4 = (i & 15) * 4;
      Bt[(n4 + 0) * LDB + kk] = (__bf16)fb[j].x;
      Bt[(n4 + 1) * LDB + kk] = (__bf16)fb[j].y;
      Bt[(n4 + 2) * LDB + kk] = (__bf16)fb[j].z;
      Bt[(n4 + 3) * LDB + kk] = (__bf16)fb[j].w;
    }
    __syncthreads();

    Frag a[2], b[2];
#pragma unroll
    for (int mi = 0; mi < 2; ++mi) {  // A frag: lane<16 -> K[0..7],[16..23]; lane>=16 -> +8
      int ar = wm + mi * 16 + lm;
      a[mi].h[0] = *(const v8bf*)&As[ar * LDA + half * 8];
      a[mi].h[1] = *(const v8bf*)&As[ar * LDA + 16 + half * 8];
    }
#pragma unroll
    for (int ni = 0; ni < 2; ++ni) {  // B frag: lane<16 -> K[0..7],[16..23] at N=lm; lane>=16 -> +8
      int br = wn + ni * 16 + lm;
      b[ni].h[0] = *(const v8bf*)&Bt[br * LDB + half * 8];
      b[ni].h[1] = *(const v8bf*)&Bt[br * LDB + 16 + half * 8];
    }
#pragma unroll
    for (int mi = 0; mi < 2; ++mi)
#pragma unroll
      for (int ni = 0; ni < 2; ++ni)
        acc[mi][ni] = wmma_bf16(a[mi], b[ni], acc[mi][ni]);
    __syncthreads();
  }
  // epilogue: +bias, tanh, store bf16. C/D layout: VGPR r -> row r+8*half, col lm.
#pragma unroll
  for (int mi = 0; mi < 2; ++mi)
#pragma unroll
    for (int ni = 0; ni < 2; ++ni) {
      int n = n0 + wn + ni * 16 + lm;
      float bv = bias[n];
#pragma unroll
      for (int r = 0; r < 8; ++r) {
        int m = m0 + wm + mi * 16 + half * 8 + r;
        out[(size_t)m * H_ + n] = (__bf16)tanhf(acc[mi][ni][r] + bv);
      }
    }
}

// ---------------------------------------------------------------------------
// Kernel 2: scores[b,t,s] = dot(decoder[b,t,:], enc[b,s,:]) + input_bias[b,s]
//           per batch: M=T=256, N=S=512, K=H=1024. enc bf16 is already N-major,
//           so the B tile is a straight copy -> use async global->LDS path.
// ---------------------------------------------------------------------------
__global__ __launch_bounds__(256) void k_scores(
    const float*  __restrict__ dec,     // [B,T,H]
    const __bf16* __restrict__ enc,     // [B,S,H] bf16
    const float*  __restrict__ in_bias, // [B,S]
    float* __restrict__ scores)         // [B,T,S]
{
  __shared__ __bf16 As[BM * LDA];
  __shared__ __bf16 Bt[BN * LDB];
  const int tid = threadIdx.x;
  const int m0  = blockIdx.x * BM;   // over T (2 tiles)
  const int n0  = blockIdx.y * BN;   // over S (8 tiles)
  const int bb  = blockIdx.z;
  const float*  Ab = dec + (size_t)bb * T_ * H_;
  const __bf16* Bb = enc + (size_t)bb * S_ * H_;
  const int wave = tid >> 5, lane = tid & 31;
  const int half = lane >> 4, lm = lane & 15;
  const int wm = (wave & 3) * 32, wn = (wave >> 2) * 32;

  // Per-thread fixed B-tile chunk: one 16B async copy per k-step.
  // Flat pointer to __shared__ carries the LDS byte offset in its low 32 bits
  // (ISA 10.2: LDS aperture).
  const int brow = tid >> 2, bc8 = (tid & 3) * 8;
  const unsigned lds_dst = (unsigned)(uintptr_t)&Bt[brow * LDB + bc8];
  const __bf16* bsrc_base = Bb + (size_t)(n0 + brow) * H_ + bc8;

  v8f acc[2][2] = {};
  for (int k0 = 0; k0 < H_; k0 += BK) {
    // --- async B-tile copy: global -> LDS, no VGPR round trip (ASYNCcnt) ---
    {
      unsigned long long gaddr = (unsigned long long)(uintptr_t)(bsrc_base + k0);
      asm volatile("global_load_async_to_lds_b128 %0, %1, off"
                   :: "v"(lds_dst), "v"(gaddr) : "memory");
    }
    // --- A tile: 128x32 f32 -> bf16, batched loads then converted stores ---
    float4 fa[4];
#pragma unroll
    for (int j = 0; j < 4; ++j) {
      int i = tid + j * 256;
      int row = i >> 3, c4 = (i & 7) * 4;
      fa[j] = *(const float4*)(Ab + (size_t)(m0 + row) * H_ + k0 + c4);
    }
#pragma unroll
    for (int j = 0; j < 4; ++j) {
      int i = tid + j * 256;
      int row = i >> 3, c4 = (i & 7) * 4;
      __bf16* d = &As[row * LDA + c4];
      d[0] = (__bf16)fa[j].x; d[1] = (__bf16)fa[j].y;
      d[2] = (__bf16)fa[j].z; d[3] = (__bf16)fa[j].w;
    }
    asm volatile("s_wait_asynccnt 0x0" ::: "memory");
    __syncthreads();

    Frag a[2], b[2];
#pragma unroll
    for (int mi = 0; mi < 2; ++mi) {
      int ar = wm + mi * 16 + lm;
      a[mi].h[0] = *(const v8bf*)&As[ar * LDA + half * 8];
      a[mi].h[1] = *(const v8bf*)&As[ar * LDA + 16 + half * 8];
    }
#pragma unroll
    for (int ni = 0; ni < 2; ++ni) {
      int br = wn + ni * 16 + lm;
      b[ni].h[0] = *(const v8bf*)&Bt[br * LDB + half * 8];
      b[ni].h[1] = *(const v8bf*)&Bt[br * LDB + 16 + half * 8];
    }
#pragma unroll
    for (int mi = 0; mi < 2; ++mi)
#pragma unroll
      for (int ni = 0; ni < 2; ++ni)
        acc[mi][ni] = wmma_bf16(a[mi], b[ni], acc[mi][ni]);
    __syncthreads();
  }
#pragma unroll
  for (int mi = 0; mi < 2; ++mi)
#pragma unroll
    for (int ni = 0; ni < 2; ++ni) {
      int s = n0 + wn + ni * 16 + lm;
      float bv = in_bias[bb * S_ + s];
#pragma unroll
      for (int r = 0; r < 8; ++r) {
        int t = m0 + wm + mi * 16 + half * 8 + r;
        scores[((size_t)bb * T_ + t) * S_ + s] = acc[mi][ni][r] + bv;
      }
    }
}

// ---------------------------------------------------------------------------
// Kernel 3: in-place softmax over S=512 per row; one 256-thread block per row
// ---------------------------------------------------------------------------
__global__ __launch_bounds__(256) void k_softmax(float* __restrict__ p) {
  __shared__ float red[256];
  float* x = p + (size_t)blockIdx.x * S_;
  const int tid = threadIdx.x;
  float a = x[tid], b = x[tid + 256];
  red[tid] = fmaxf(a, b);
  __syncthreads();
  for (int o = 128; o > 0; o >>= 1) {
    if (tid < o) red[tid] = fmaxf(red[tid], red[tid + o]);
    __syncthreads();
  }
  float mx = red[0];
  __syncthreads();
  float e0 = __expf(a - mx), e1 = __expf(b - mx);
  red[tid] = e0 + e1;
  __syncthreads();
  for (int o = 128; o > 0; o >>= 1) {
    if (tid < o) red[tid] += red[tid + o];
    __syncthreads();
  }
  float inv = 1.0f / red[0];
  x[tid] = e0 * inv;
  x[tid + 256] = e1 * inv;
}

// ---------------------------------------------------------------------------
// Kernel 4: zero-fill output (131 MB) with float4 stores — the HBM floor
// ---------------------------------------------------------------------------
__global__ __launch_bounds__(256) void k_zero(float4* __restrict__ o) {
  size_t i = (size_t)blockIdx.x * 256 + threadIdx.x;
  o[i] = make_float4(0.f, 0.f, 0.f, 0.f);
}

// ---------------------------------------------------------------------------
// Kernel 5: scatter-add probs[b,t,s] into out[b,t, inputs[b,s]]
// ---------------------------------------------------------------------------
__global__ __launch_bounds__(256) void k_scatter(
    const float* __restrict__ probs,  // [B,T,S]
    const int*   __restrict__ ids,    // [B,S]
    float* __restrict__ out)          // [B,T,V]
{
  int idx = blockIdx.x * 256 + threadIdx.x;  // enumerates (b,t,s)
  int s  = idx & (S_ - 1);
  int bt = idx >> 9;        // b*T + t
  int b  = bt >> 8;         // T = 256
  float v = probs[idx];
  int id = ids[b * S_ + s];
  atomicAdd(out + (size_t)bt * V_ + id, v);
}

extern "C" void kernel_launch(void* const* d_in, const int* in_sizes, int n_in,
                              void* d_out, int out_size, void* d_ws, size_t ws_size,
                              hipStream_t stream) {
  const int*   inputs = (const int*)d_in[0];    // [B,S] int32
  const float* enc_f  = (const float*)d_in[1];  // [B,S,H]
  const float* dec    = (const float*)d_in[2];  // [B,T,H]
  const float* ibias  = (const float*)d_in[3];  // [B,S]
  const float* W      = (const float*)d_in[4];  // [H,H]
  const float* bproj  = (const float*)d_in[5];  // [H]
  float* out = (float*)d_out;

  // workspace: enc bf16 (4 MiB) | scores/probs f32 (2 MiB)
  __bf16* encb  = (__bf16*)d_ws;
  float* scores = (float*)((char*)d_ws + (size_t)B_ * S_ * H_ * sizeof(__bf16));

  (void)in_sizes; (void)n_in; (void)out_size; (void)ws_size;

  // 1) enc = tanh(encoder @ W + b)  -> bf16
  k_gemm_tanh<<<dim3((B_ * S_) / BM, H_ / BN), 256, 0, stream>>>(enc_f, W, bproj, encb);
  // 2) scores = decoder @ enc^T + input_bias (batched)
  k_scores<<<dim3(T_ / BM, S_ / BN, B_), 256, 0, stream>>>(dec, encb, ibias, scores);
  // 3) softmax over S, in place
  k_softmax<<<B_ * T_, 256, 0, stream>>>(scores);
  // 4) zero out [B,T,V]
  k_zero<<<(B_ * T_ * V_ / 4) / 256, 256, 0, stream>>>((float4*)out);
  // 5) scatter-add into vocab slots
  k_scatter<<<(B_ * T_ * S_) / 256, 256, 0, stream>>>(scores, inputs, out);
}
